// VectorQuantizer_7275674599497
// MI455X (gfx1250) — matrix operand order repaired and checked by reading, not compiled, not run on hardware
//
#include <hip/hip_runtime.h>
#include <hip/hip_bf16.h>

typedef unsigned short ushort_t;
typedef __bf16 v16bf __attribute__((ext_vector_type(16)));
typedef float  v8f   __attribute__((ext_vector_type(8)));

#define N_TOK 8192   // B*S
#define KC    8192   // codebook size
#define DIM   512
#define TOK_PER_WG 64
#define N_WG  (N_TOK / TOK_PER_WG)   // 128
#define INV_TEMP 100.0f              // 1/ENT_TEMP

union Frag { uint4 q[2]; v16bf v; };

__device__ inline float waveReduceAdd(float v) {
#pragma unroll
  for (int m = 16; m >= 1; m >>= 1) v += __shfl_xor(v, m, 32);
  return v;
}

// B fragment: 32x16 bf16 tile column, lane holds code column, contiguous 16 K
__device__ inline Frag loadB(const ushort_t* __restrict__ cbf, size_t bRow,
                             int s, int half) {
  Frag f;
  const uint4* p = reinterpret_cast<const uint4*>(cbf + bRow + s * 32 + half * 16);
  f.q[0] = p[0];
  f.q[1] = p[1];
  return f;
}

// ---------------------------------------------------------------------------
// Kernel 1: convert x / codebook rows to bf16 (RNE) and compute row ||.||^2
// ---------------------------------------------------------------------------
__global__ __launch_bounds__(128) void vq_prep(const float* __restrict__ x,
                                               const float* __restrict__ cb,
                                               ushort_t* __restrict__ xbf,
                                               ushort_t* __restrict__ cbf,
                                               float* __restrict__ xsq,
                                               float* __restrict__ csq) {
  int row = blockIdx.x;
  const float* src;
  ushort_t* dst;
  float* sq;
  int r;
  if (row < N_TOK) {
    r = row; src = x + (size_t)r * DIM; dst = xbf + (size_t)r * DIM; sq = xsq;
  } else {
    r = row - N_TOK; src = cb + (size_t)r * DIM; dst = cbf + (size_t)r * DIM; sq = csq;
  }
  float acc = 0.f;
  for (int j = threadIdx.x; j < DIM; j += 128) {
    float v = src[j];
    acc += v * v;
    unsigned u = __float_as_uint(v);
    unsigned rnd = u + 0x7fffu + ((u >> 16) & 1u);   // round-to-nearest-even bf16
    dst[j] = (ushort_t)(rnd >> 16);
  }
  acc = waveReduceAdd(acc);
  __shared__ float p[4];
  if ((threadIdx.x & 31) == 0) p[threadIdx.x >> 5] = acc;
  __syncthreads();
  if (threadIdx.x == 0) sq[r] = p[0] + p[1] + p[2] + p[3];
}

// ---------------------------------------------------------------------------
// Kernel 2: main fused GEMM (bf16 WMMA) + online softmax + argmin (pass 1),
//           then per-k probability sums (pass 2). One WG owns 64 tokens and
//           sweeps all K; wave w covers codes {w*16 + 64*i + (lane&15)}.
//           B fragments are software-pipelined 4 steps deep so ~8 b128 loads
//           stay in flight (needed: A tile takes 64KB LDS -> 1 wave/SIMD,
//           so in-stream pipelining is the only latency hiding available).
// ---------------------------------------------------------------------------
__global__ __launch_bounds__(128) void vq_main(const ushort_t* __restrict__ xbf,
                                               const ushort_t* __restrict__ cbf,
                                               const float* __restrict__ xsq,
                                               const float* __restrict__ csq,
                                               int* __restrict__ out_idx,
                                               float* __restrict__ ws_avg,
                                               float* __restrict__ ws_H) {
  __shared__ ushort_t sA[TOK_PER_WG * DIM];            // 64KB A tile (bf16)
  __shared__ float sLogZ[TOK_PER_WG];
  __shared__ float sH[TOK_PER_WG];
  __shared__ float pM[TOK_PER_WG][4], pS[TOK_PER_WG][4], pT[TOK_PER_WG][4], pD[TOK_PER_WG][4];
  __shared__ int   pI[TOK_PER_WG][4];

  const int tid  = threadIdx.x;
  const int wg   = blockIdx.x;
  const int w    = tid >> 5;       // wave id 0..3
  const int ln   = tid & 31;
  const int col  = ln & 15;        // N column within 16x16 tile
  const int half = ln >> 4;        // 0: rows 0..7 / K-lo chunks, 1: rows 8..15 / K-hi
  const int tokBase = wg * TOK_PER_WG;

  // Stage A tile into LDS (64 rows x 512 bf16)
  {
    const uint4* src = reinterpret_cast<const uint4*>(xbf + (size_t)tokBase * DIM);
    uint4* dst = reinterpret_cast<uint4*>(sA);
    for (int i = tid; i < TOK_PER_WG * DIM / 8; i += 128) dst[i] = src[i];
  }
  __syncthreads();

  // Preload ||x||^2 for my 32 (tile,row) slots
  float xq[32];
#pragma unroll
  for (int mt = 0; mt < 4; ++mt)
#pragma unroll
    for (int r = 0; r < 8; ++r)
      xq[mt * 8 + r] = xsq[tokBase + mt * 16 + half * 8 + r];

  float mn2[32], mM[32], mS[32], mT[32];
  int ai[32];
#pragma unroll
  for (int i = 0; i < 32; ++i) {
    mn2[i] = 3.4e38f; ai[i] = 0; mM[i] = -3.4e38f; mS[i] = 0.f; mT[i] = 0.f;
  }

  // ---------------- Pass 1: distances -> argmin + online softmax stats ------
  Frag Bb[4];
  {
    const size_t r0 = (size_t)(w * 16 + col) * DIM;
#pragma unroll
    for (int p = 0; p < 4; ++p) Bb[p] = loadB(cbf, r0, p, half);
  }
#pragma unroll 1
  for (int kb = w * 16; kb < KC; kb += 64) {
    const int kcode = kb + col;
    const size_t bRow = (size_t)kcode * DIM;
    const size_t bRowN = (size_t)((kb + 64 < KC) ? (kcode + 64) : kcode) * DIM;

    v8f acc[4];
#pragma unroll
    for (int mt = 0; mt < 4; ++mt) acc[mt] = v8f{0.f,0.f,0.f,0.f,0.f,0.f,0.f,0.f};

#pragma unroll
    for (int s = 0; s < 16; ++s) {
#pragma unroll
      for (int mt = 0; mt < 4; ++mt) {
        // A fragment: 16x32 bf16, lane row = col, contiguous 16 K per half, same order as B
        Frag A;
        const ushort_t* arow = sA + (mt * 16 + col) * DIM + s * 32 + half * 16;
        A.q[0] = *reinterpret_cast<const uint4*>(arow);
        A.q[1] = *reinterpret_cast<const uint4*>(arow + 8);
        acc[mt] = __builtin_amdgcn_wmma_f32_16x16x32_bf16(
            false, A.v, false, Bb[s & 3].v, (short)0, acc[mt], false, false);
      }
      // prefetch fragment s+4 (rolls into the next kb iteration's steps 0..3)
      Bb[s & 3] = (s < 12) ? loadB(cbf, bRow, s + 4, half)
                           : loadB(cbf, bRowN, s - 12, half);
    }

    const float cs = csq[kcode];
#pragma unroll
    for (int mt = 0; mt < 4; ++mt)
#pragma unroll
      for (int r = 0; r < 8; ++r) {
        const int i = mt * 8 + r;
        float dot = acc[mt][r];
        float d2 = fmaxf(xq[i] - 2.f * dot + cs, 0.f);
        if (d2 < mn2[i]) { mn2[i] = d2; ai[i] = kcode; }
        float l = -INV_TEMP * sqrtf(d2);
        float nm = fmaxf(mM[i], l);
        float sc = __expf(mM[i] - nm);
        float e  = __expf(l - nm);
        mS[i] = mS[i] * sc + e;
        mT[i] = mT[i] * sc + e * l;
        mM[i] = nm;
      }
  }

  // Reduce across the 16 lane-columns within each half (wave32 shuffles)
#pragma unroll
  for (int mask = 1; mask <= 8; mask <<= 1) {
#pragma unroll
    for (int i = 0; i < 32; ++i) {
      float od = __shfl_xor(mn2[i], mask, 32);
      int   oi = __shfl_xor(ai[i],  mask, 32);
      if (od < mn2[i] || (od == mn2[i] && oi < ai[i])) { mn2[i] = od; ai[i] = oi; }
      float om = __shfl_xor(mM[i], mask, 32);
      float os = __shfl_xor(mS[i], mask, 32);
      float ot = __shfl_xor(mT[i], mask, 32);
      float nm = fmaxf(mM[i], om);
      float s1 = __expf(mM[i] - nm), s2 = __expf(om - nm);
      mS[i] = mS[i] * s1 + os * s2;
      mT[i] = mT[i] * s1 + ot * s2;
      mM[i] = nm;
    }
  }
  if (col == 0) {
#pragma unroll
    for (int mt = 0; mt < 4; ++mt)
#pragma unroll
      for (int r = 0; r < 8; ++r) {
        const int row = mt * 16 + half * 8 + r;
        const int i = mt * 8 + r;
        pM[row][w] = mM[i]; pS[row][w] = mS[i]; pT[row][w] = mT[i];
        pD[row][w] = mn2[i]; pI[row][w] = ai[i];
      }
  }
  __syncthreads();

  // Combine the 4 wave partials per token
  if (tid < TOK_PER_WG) {
    float bm = -3.4e38f, bs = 0.f, bt = 0.f, bd = 3.4e38f;
    int bi = 0x7fffffff;
    for (int j = 0; j < 4; ++j) {
      float om = pM[tid][j], os = pS[tid][j], ot = pT[tid][j];
      float nm = fmaxf(bm, om);
      float s1 = __expf(bm - nm), s2 = __expf(om - nm);
      bs = bs * s1 + os * s2;
      bt = bt * s1 + ot * s2;
      bm = nm;
      float od = pD[tid][j]; int oi = pI[tid][j];
      if (od < bd || (od == bd && oi < bi)) { bd = od; bi = oi; }
    }
    float lz = bm + __logf(bs);
    sLogZ[tid] = lz;
    sH[tid] = lz - bt / bs;        // per-token sample entropy
    out_idx[tokBase + tid] = bi;
  }
  __syncthreads();
  if (tid == 0) {
    float h = 0.f;
    for (int j = 0; j < TOK_PER_WG; ++j) h += sH[j];
    ws_H[wg] = h;                  // deterministic per-WG partial
  }

  // Preload logZ for my 32 slots
  float lzr[32];
#pragma unroll
  for (int mt = 0; mt < 4; ++mt)
#pragma unroll
    for (int r = 0; r < 8; ++r)
      lzr[mt * 8 + r] = sLogZ[mt * 16 + half * 8 + r];

  // ---------------- Pass 2: per-k probability sums (avg_probs partials) -----
  {
    const size_t r0 = (size_t)(w * 16 + col) * DIM;
#pragma unroll
    for (int p = 0; p < 4; ++p) Bb[p] = loadB(cbf, r0, p, half);
  }
#pragma unroll 1
  for (int kb = w * 16; kb < KC; kb += 64) {
    const int kcode = kb + col;
    const size_t bRow = (size_t)kcode * DIM;
    const size_t bRowN = (size_t)((kb + 64 < KC) ? (kcode + 64) : kcode) * DIM;

    v8f acc[4];
#pragma unroll
    for (int mt = 0; mt < 4; ++mt) acc[mt] = v8f{0.f,0.f,0.f,0.f,0.f,0.f,0.f,0.f};

#pragma unroll
    for (int s = 0; s < 16; ++s) {
#pragma unroll
      for (int mt = 0; mt < 4; ++mt) {
        Frag A;
        const ushort_t* arow = sA + (mt * 16 + col) * DIM + s * 32 + half * 16;
        A.q[0] = *reinterpret_cast<const uint4*>(arow);
        A.q[1] = *reinterpret_cast<const uint4*>(arow + 8);
        acc[mt] = __builtin_amdgcn_wmma_f32_16x16x32_bf16(
            false, A.v, false, Bb[s & 3].v, (short)0, acc[mt], false, false);
      }
      Bb[s & 3] = (s < 12) ? loadB(cbf, bRow, s + 4, half)
                           : loadB(cbf, bRowN, s - 12, half);
    }

    const float cs = csq[kcode];
    float psum = 0.f;
#pragma unroll
    for (int mt = 0; mt < 4; ++mt)
#pragma unroll
      for (int r = 0; r < 8; ++r) {
        const int i = mt * 8 + r;
        float dot = acc[mt][r];
        float d2 = fmaxf(xq[i] - 2.f * dot + cs, 0.f);
        float l = -INV_TEMP * sqrtf(d2);
        psum += __expf(l - lzr[i]);
      }
    psum += __shfl_xor(psum, 16, 32);          // fold the two row-halves
    if (half == 0)                              // exactly one lane owns each k
      ws_avg[(size_t)wg * KC + kcode] = psum;   // deterministic, no atomics
  }
}

// ---------------------------------------------------------------------------
// Kernel 3: gather quantized vectors, straight-through output, MSE partials
// ---------------------------------------------------------------------------
__global__ __launch_bounds__(128) void vq_quant(const float* __restrict__ x,
                                                const float* __restrict__ cb,
                                                const int* __restrict__ idx,
                                                float* __restrict__ outQ,
                                                float* __restrict__ outIdxF,
                                                float* __restrict__ ws_sq) {
  const int t = blockIdx.x;
  const int k = idx[t];
  const float* xr = x + (size_t)t * DIM;
  const float* cr = cb + (size_t)k * DIM;
  float acc = 0.f;
  for (int j = threadIdx.x; j < DIM; j += 128) {
    float xv = xr[j], q = cr[j];
    outQ[(size_t)t * DIM + j] = xv + (q - xv);   // straight-through forward
    float d = xv - q;
    acc += d * d;
  }
  acc = waveReduceAdd(acc);
  __shared__ float p[4];
  if ((threadIdx.x & 31) == 0) p[threadIdx.x >> 5] = acc;
  __syncthreads();
  if (threadIdx.x == 0) {
    ws_sq[t] = p[0] + p[1] + p[2] + p[3];
    outIdxF[t] = (float)k;
  }
}

// ---------------------------------------------------------------------------
// Kernel 4: reduce ws_avg over WGs per k -> avg-entropy partial per block
// ---------------------------------------------------------------------------
__global__ __launch_bounds__(256) void vq_avg(const float* __restrict__ ws_avg,
                                              float* __restrict__ entpart) {
  const int k = blockIdx.x * 256 + threadIdx.x;
  float s = 0.f;
  for (int wgi = 0; wgi < N_WG; ++wgi) s += ws_avg[(size_t)wgi * KC + k];
  float ap = s * (1.0f / (float)N_TOK);
  float c = -ap * __logf(ap + 1e-8f);
  c = waveReduceAdd(c);
  __shared__ float p[8];
  if ((threadIdx.x & 31) == 0) p[threadIdx.x >> 5] = c;
  __syncthreads();
  if (threadIdx.x == 0) {
    float t = 0.f;
    for (int j = 0; j < 8; ++j) t += p[j];
    entpart[blockIdx.x] = t;
  }
}

// ---------------------------------------------------------------------------
// Kernel 5: final scalars
// ---------------------------------------------------------------------------
__global__ __launch_bounds__(256) void vq_final(const float* __restrict__ entpart,
                                                const float* __restrict__ ws_H,
                                                const float* __restrict__ ws_sq,
                                                float* __restrict__ outs) {
  float a = 0.f;
  for (int j = threadIdx.x; j < N_TOK; j += 256) a += ws_sq[j];
  a = waveReduceAdd(a);
  __shared__ float p[8];
  if ((threadIdx.x & 31) == 0) p[threadIdx.x >> 5] = a;
  __syncthreads();
  if (threadIdx.x == 0) {
    float ssq = 0.f;
    for (int j = 0; j < 8; ++j) ssq += p[j];
    float sH = 0.f;
    for (int j = 0; j < N_WG; ++j) sH += ws_H[j];
    float aE = 0.f;
    for (int j = 0; j < KC / 256; ++j) aE += entpart[j];
    float sampleH = sH * (1.0f / (float)N_TOK);
    float ent = (sampleH - aE) * 0.1f;                    // ENT_W
    float M = 0.5f * ssq * (1.0f / ((float)N_TOK * (float)DIM));
    outs[0] = M + 0.25f * M + ent;   // loss
    outs[1] = 0.25f * M;             // commit_loss (COMMIT_W * M)
    outs[2] = M;                     // codebook_loss
    outs[3] = ent;                   // entropy_loss
  }
}

// ---------------------------------------------------------------------------
extern "C" void kernel_launch(void* const* d_in, const int* in_sizes, int n_in,
                              void* d_out, int out_size, void* d_ws, size_t ws_size,
                              hipStream_t stream) {
  (void)in_sizes; (void)n_in; (void)out_size; (void)ws_size;
  const float* x  = (const float*)d_in[0];   // (B,S,D) fp32
  const float* cb = (const float*)d_in[1];   // (K,D)  fp32
  float* out = (float*)d_out;                // [quantized | 4 scalars | indices]

  char* ws = (char*)d_ws;                    // needs ~20.1 MB scratch
  size_t o = 0;
  ushort_t* xbf = (ushort_t*)(ws + o); o += (size_t)N_TOK * DIM * 2;   // 8 MB
  ushort_t* cbf = (ushort_t*)(ws + o); o += (size_t)KC    * DIM * 2;   // 8 MB
  float* xsq    = (float*)(ws + o);    o += (size_t)N_TOK * 4;
  float* csq    = (float*)(ws + o);    o += (size_t)KC    * 4;
  int*   idx    = (int*)(ws + o);      o += (size_t)N_TOK * 4;
  float* ws_avg = (float*)(ws + o);    o += (size_t)N_WG * KC * 4;     // 4 MB
  float* ws_H   = (float*)(ws + o);    o += (size_t)N_WG * 4;
  float* ws_sq  = (float*)(ws + o);    o += (size_t)N_TOK * 4;
  float* entpart= (float*)(ws + o);    o += (size_t)(KC / 256) * 4;

  float* outQ    = out;
  float* outScal = out + (size_t)N_TOK * DIM;
  float* outIdxF = out + (size_t)N_TOK * DIM + 4;

  vq_prep <<<N_TOK + KC, 128, 0, stream>>>(x, cb, xbf, cbf, xsq, csq);
  vq_main <<<N_WG,       128, 0, stream>>>(xbf, cbf, xsq, csq, idx, ws_avg, ws_H);
  vq_quant<<<N_TOK,      128, 0, stream>>>(x, cb, idx, outQ, outIdxF, ws_sq);
  vq_avg  <<<KC / 256,   256, 0, stream>>>(ws_avg, entpart);
  vq_final<<<1,          256, 0, stream>>>(entpart, ws_H, ws_sq, outScal);
}
